// ProbLinear_9577777070516
// MI455X (gfx1250) — matrix operand order repaired and checked by reading, not compiled, not run on hardware
//
#include <hip/hip_runtime.h>
#include <math.h>

// ---- problem sizes (from reference) ----
#define IN_F   4096
#define OUT_F  4096
#define TOKENS 8192

// ---- GEMM tiling ----
#define BM 128
#define BN 128
#define BK 32
#define NKT (IN_F / BK)   // 128 K-tiles
#define LDSA 40           // padded row stride (bf16 units) for A tile
#define LDSB 40           // padded row stride (bf16 units) for B tile

typedef __attribute__((ext_vector_type(16))) __bf16 v16bf;
typedef __attribute__((ext_vector_type(8)))  __bf16 v8bf;
typedef __attribute__((ext_vector_type(4)))  __bf16 v4bf;
typedef __attribute__((ext_vector_type(8)))  float  v8f;

__device__ __forceinline__ float softplusf(float r) {
    // numerically stable log1p(exp(r))
    return fmaxf(r, 0.0f) + log1pf(expf(-fabsf(r)));
}

__device__ __forceinline__ void split_bf16(float w, __bf16& hi, __bf16& lo) {
    hi = (__bf16)w;
    lo = (__bf16)(w - (float)hi);
}

// CDNA5 async copy: global -> LDS, 16 bytes per lane, tracked by ASYNCcnt.
// GVS addressing: uniform 64-bit SGPR base + per-lane 32-bit byte offset.
__device__ __forceinline__ void async_b128_to_lds(unsigned lds_off, unsigned voff,
                                                  unsigned long long sbase) {
    asm volatile("global_load_async_to_lds_b128 %0, %1, %2"
                 :: "v"(lds_off), "v"(voff), "s"(sbase) : "memory");
}

__device__ __forceinline__ void wait_async0() {
    asm volatile("s_wait_asynccnt 0x0" ::: "memory");
}

// low 32 bits of a generic pointer into LDS == LDS byte offset
__device__ __forceinline__ unsigned lds_off32(const void* p) {
    return (unsigned)(size_t)p;
}

// ---------------------------------------------------------------------------
// Kernel 1: sample W = mu + softplus(rho)*eps, split into bf16 hi/lo planes.
// ---------------------------------------------------------------------------
__global__ __launch_bounds__(256) void sample_weight_split(
    const float* __restrict__ mu, const float* __restrict__ rho,
    const float* __restrict__ eps,
    __bf16* __restrict__ whi, __bf16* __restrict__ wlo)
{
    size_t i = ((size_t)blockIdx.x * 256 + threadIdx.x) * 4;
    float4 m = *(const float4*)(mu  + i);
    float4 r = *(const float4*)(rho + i);
    float4 e = *(const float4*)(eps + i);
    float w0 = m.x + softplusf(r.x) * e.x;
    float w1 = m.y + softplusf(r.y) * e.y;
    float w2 = m.z + softplusf(r.z) * e.z;
    float w3 = m.w + softplusf(r.w) * e.w;
    v4bf hi, lo;
    { __bf16 h, l; split_bf16(w0, h, l); hi[0]=h; lo[0]=l; }
    { __bf16 h, l; split_bf16(w1, h, l); hi[1]=h; lo[1]=l; }
    { __bf16 h, l; split_bf16(w2, h, l); hi[2]=h; lo[2]=l; }
    { __bf16 h, l; split_bf16(w3, h, l); hi[3]=h; lo[3]=l; }
    *(v4bf*)(whi + i) = hi;
    *(v4bf*)(wlo + i) = lo;
}

// ---------------------------------------------------------------------------
// Kernel 2: sample bias (f32)
// ---------------------------------------------------------------------------
__global__ __launch_bounds__(256) void sample_bias(
    const float* __restrict__ bmu, const float* __restrict__ brho,
    const float* __restrict__ beps, float* __restrict__ bout)
{
    int i = blockIdx.x * 256 + threadIdx.x;
    if (i < OUT_F) bout[i] = bmu[i] + softplusf(brho[i]) * beps[i];
}

// ---------------------------------------------------------------------------
// Kernel 3: out[t,o] = sum_i x[t,i]*W[o,i] + b[o]
// Split-bf16 WMMA GEMM with double-buffered LDS:
//   A tile: global f32 -> hi/lo bf16 split -> ds_store   (DScnt)
//   B tile: global_load_async_to_lds_b128 copies         (ASYNCcnt)
// 8 wave32s arranged 2(M) x 4(N); wave tile 64x32; 24 WMMAs per K-step.
// ---------------------------------------------------------------------------
__global__ __launch_bounds__(256) void gemm_wmma_bf16x3(
    const float*  __restrict__ x,
    const __bf16* __restrict__ whi,
    const __bf16* __restrict__ wlo,
    const float*  __restrict__ bias,
    float* __restrict__ out)
{
    __shared__ __bf16 sAhi[2][BM * LDSA];
    __shared__ __bf16 sAlo[2][BM * LDSA];
    __shared__ __bf16 sBhi[2][BN * LDSB];
    __shared__ __bf16 sBlo[2][BN * LDSB];

    const int tid  = threadIdx.x;
    const int lane = tid & 31;
    const int wid  = tid >> 5;
    const int wm   = wid & 1;        // wave M group: 0..1 (64 rows each)
    const int wn   = wid >> 1;       // wave N group: 0..3 (32 cols each)
    const int t0   = blockIdx.y * BM;
    const int o0   = blockIdx.x * BN;

    const int ln = lane & 15;        // M (for A) / N (for B,C) within 16
    const int lh = lane >> 4;        // lane half selects K/M sub-range

    // A staging coords: 4 passes of 32 rows x 8 float4-chunks
    const int arow = tid >> 3;             // 0..31
    const int acol = (tid & 7) * 4;        // 0..28 (f32 cols)
    // B async-copy coords: 2 passes; chunk = 8 bf16 (16 B), 4 chunks/row
    const int brow0 = tid >> 2;            // 0..63
    const int bcol  = (tid & 3) * 8;       // bf16 col offset

    v8f acc[4][2];
    #pragma unroll
    for (int m = 0; m < 4; ++m)
        #pragma unroll
        for (int n = 0; n < 2; ++n)
            #pragma unroll
            for (int r = 0; r < 8; ++r)
                acc[m][n][r] = 0.0f;

    union Frag { v16bf v; v8bf h[2]; };

    // per-thread global byte offsets for B copies (loop invariant)
    unsigned bvoff[2];
    #pragma unroll
    for (int p = 0; p < 2; ++p) {
        int row = brow0 + p * 64;
        bvoff[p] = (unsigned)((((o0 + row) * IN_F) + bcol) * 2);
    }

    // ---- stage helpers (inlined manually via lambdas) ----
    auto stage_A = [&](int k0, int buf) {
        #pragma unroll
        for (int p = 0; p < 4; ++p) {
            int r = arow + p * 32;
            float4 v = *(const float4*)(x + (size_t)(t0 + r) * IN_F + k0 + acol);
            v4bf hi, lo;
            { __bf16 h, l; split_bf16(v.x, h, l); hi[0]=h; lo[0]=l; }
            { __bf16 h, l; split_bf16(v.y, h, l); hi[1]=h; lo[1]=l; }
            { __bf16 h, l; split_bf16(v.z, h, l); hi[2]=h; lo[2]=l; }
            { __bf16 h, l; split_bf16(v.w, h, l); hi[3]=h; lo[3]=l; }
            *(v4bf*)(&sAhi[buf][r * LDSA + acol]) = hi;
            *(v4bf*)(&sAlo[buf][r * LDSA + acol]) = lo;
        }
    };
    auto issue_B = [&](int k0, int buf) {
        unsigned long long basehi = (unsigned long long)(size_t)(whi + k0);
        unsigned long long baselo = (unsigned long long)(size_t)(wlo + k0);
        #pragma unroll
        for (int p = 0; p < 2; ++p) {
            int row = brow0 + p * 64;
            unsigned dhi = lds_off32(&sBhi[buf][row * LDSB + bcol]);
            unsigned dlo = lds_off32(&sBlo[buf][row * LDSB + bcol]);
            async_b128_to_lds(dhi, bvoff[p], basehi);
            async_b128_to_lds(dlo, bvoff[p], baselo);
        }
    };

    // ---- prologue: stage tile 0 into buffer 0 ----
    stage_A(0, 0);
    issue_B(0, 0);

    for (int kt = 0; kt < NKT; ++kt) {
        const int cur = kt & 1;
        wait_async0();        // B tile `kt` (this wave's share) landed in LDS
        __syncthreads();      // all waves' A stores + B asyncs visible

        if (kt + 1 < NKT) {   // prefetch next tile into the other buffer
            stage_A((kt + 1) * BK, cur ^ 1);
            issue_B((kt + 1) * BK, cur ^ 1);
        }

        // ---- B fragments: lane ln = column N, lanes 16-31 carry K=16..31 ----
        Frag bhiF[2], bloF[2];
        #pragma unroll
        for (int n = 0; n < 2; ++n) {
            const __bf16* p0 = &sBhi[cur][(wn * 32 + n * 16 + ln) * LDSB + lh * 16];
            bhiF[n].h[0] = *(const v8bf*)p0;
            bhiF[n].h[1] = *(const v8bf*)(p0 + 8);
            const __bf16* p1 = &sBlo[cur][(wn * 32 + n * 16 + ln) * LDSB + lh * 16];
            bloF[n].h[0] = *(const v8bf*)p1;
            bloF[n].h[1] = *(const v8bf*)(p1 + 8);
        }

        // ---- A fragments + WMMA: lanes0-15 K0..15, lanes16-31 K16..31 ----
        #pragma unroll
        for (int m = 0; m < 4; ++m) {
            Frag ahiF, aloF;
            const __bf16* pa = &sAhi[cur][(wm * 64 + m * 16 + ln) * LDSA + lh * 16];
            ahiF.h[0] = *(const v8bf*)pa;
            ahiF.h[1] = *(const v8bf*)(pa + 8);
            const __bf16* pb = &sAlo[cur][(wm * 64 + m * 16 + ln) * LDSA + lh * 16];
            aloF.h[0] = *(const v8bf*)pb;
            aloF.h[1] = *(const v8bf*)(pb + 8);
            #pragma unroll
            for (int n = 0; n < 2; ++n) {
                acc[m][n] = __builtin_amdgcn_wmma_f32_16x16x32_bf16(
                    false, ahiF.v, false, bhiF[n].v, (short)0, acc[m][n], false, false);
                acc[m][n] = __builtin_amdgcn_wmma_f32_16x16x32_bf16(
                    false, ahiF.v, false, bloF[n].v, (short)0, acc[m][n], false, false);
                acc[m][n] = __builtin_amdgcn_wmma_f32_16x16x32_bf16(
                    false, aloF.v, false, bhiF[n].v, (short)0, acc[m][n], false, false);
            }
        }
        // no trailing barrier: next iteration's wait+barrier separates the
        // LDS reads above from the following tile's writes into this buffer
    }

    // ---- epilogue: C layout = VGPR r -> M = lh*8 + r, N = ln ----
    #pragma unroll
    for (int n = 0; n < 2; ++n) {
        int col = o0 + wn * 32 + n * 16 + ln;
        float bv = bias[col];
        #pragma unroll
        for (int m = 0; m < 4; ++m) {
            int rbase = t0 + wm * 64 + m * 16 + lh * 8;
            #pragma unroll
            for (int r = 0; r < 8; ++r) {
                out[(size_t)(rbase + r) * OUT_F + col] = acc[m][n][r] + bv;
            }
        }
    }
}

// ---------------------------------------------------------------------------
extern "C" void kernel_launch(void* const* d_in, const int* in_sizes, int n_in,
                              void* d_out, int out_size, void* d_ws, size_t ws_size,
                              hipStream_t stream) {
    const float* x    = (const float*)d_in[0];
    const float* wmu  = (const float*)d_in[1];
    const float* wrho = (const float*)d_in[2];
    const float* bmu  = (const float*)d_in[3];
    const float* brho = (const float*)d_in[4];
    const float* ew   = (const float*)d_in[5];
    const float* eb   = (const float*)d_in[6];
    float* out = (float*)d_out;

    // workspace: W_hi bf16 [4096x4096] | W_lo bf16 [4096x4096] | bias f32 [4096]
    __bf16* whi  = (__bf16*)d_ws;
    __bf16* wlo  = whi + (size_t)OUT_F * IN_F;
    float*  bsmp = (float*)(wlo + (size_t)OUT_F * IN_F);

    const int nW = (int)(((size_t)OUT_F * IN_F) / (256 * 4));
    sample_weight_split<<<nW, 256, 0, stream>>>(wmu, wrho, ew, whi, wlo);
    sample_bias<<<OUT_F / 256, 256, 0, stream>>>(bmu, brho, eb, bsmp);

    dim3 grid(OUT_F / BN, TOKENS / BM);
    gemm_wmma_bf16x3<<<grid, 256, 0, stream>>>(x, whi, wlo, bsmp, out);
}
